// LinearAttention_13237089206785
// MI455X (gfx1250) — matrix operand order repaired and checked by reading, not compiled, not run on hardware
//
#include <hip/hip_runtime.h>
#include <hip/hip_bf16.h>

// Linear attention, B=4 S=4096 D=1024 H=16 HD=64.
// All GEMMs via v_wmma_f32_16x16x32_bf16 (bf16 inputs, f32 accumulate).
// Layouts chosen so every WMMA fragment is 2x ds_load_b128 per lane.

#define BB 4
#define SS 4096
#define DD 1024
#define HH 16
#define HD 64
#define NTOK (BB * SS)          // 16384
#define EPSV 1e-6f

typedef __attribute__((ext_vector_type(16))) __bf16 v16bf;
typedef __attribute__((ext_vector_type(8)))  float  v8f;
typedef __attribute__((ext_vector_type(4)))  float  f4;
typedef __attribute__((ext_vector_type(4)))  __bf16 v4bf;

// ---------------- WMMA fragment helpers (wave32, ISA 7.12.2 layouts) -------

// A fragment 16x32 (MxK) from row-major tile t[rows][ld], rows rb..rb+15.
// Per lane: elements e=0..7 -> K=8g..8g+7 (16B run), e=8..15 -> K=16+8g..
// (second 16B run) => compiler emits 2x ds_load_b128.
__device__ __forceinline__ v16bf frag_a_rm(const __bf16* t, int ld, int rb, int lane) {
  const int g = (lane >> 4) & 1, m = lane & 15;
  v16bf a;
#pragma unroll
  for (int e = 0; e < 16; ++e) {
    const int k = e + 8 * g + ((e >= 8) ? 8 : 0);
    a[e] = t[(rb + m) * ld + k];
  }
  return a;
}

// B fragment 32x16 (KxN) from COLUMN-major tile t[n][k] (ld = K extent).
// Lane n reads K = 8g..8g+7 and 16+8g..16+8g+7 (same as A) => 2x ds_load_b128.
__device__ __forceinline__ v16bf frag_b_cm(const __bf16* t, int ld, int cb, int lane) {
  const int g = (lane >> 4) & 1, n = lane & 15;
  v16bf b;
#pragma unroll
  for (int e = 0; e < 16; ++e) {
    b[e] = t[(cb + n) * ld + e + 8 * g + ((e >= 8) ? 8 : 0)];
  }
  return b;
}

__device__ __forceinline__ v8f wmma_bf16(v16bf a, v16bf b, v8f c) {
  return __builtin_amdgcn_wmma_f32_16x16x32_bf16(false, a, false, b, (short)0, c,
                                                 false, false);
}

// ---------------- Kernel 0a: fp32 -> bf16 convert (vectorized) -------------

__global__ __launch_bounds__(256) void cvt_bf16_kernel(const float* __restrict__ in,
                                                       __bf16* __restrict__ out, int n4) {
  int i = blockIdx.x * blockDim.x + threadIdx.x;
  const int stride = gridDim.x * blockDim.x;
  const f4* I = (const f4*)in;
  v4bf* O = (v4bf*)out;
  for (; i < n4; i += stride) {
    const f4 v = I[i];
    v4bf o;
    o[0] = (__bf16)v[0]; o[1] = (__bf16)v[1];
    o[2] = (__bf16)v[2]; o[3] = (__bf16)v[3];
    O[i] = o;
  }
}

// ---------------- Kernel 0b: fp32 [K][N] -> bf16 transposed [N][K] ---------
// 64x64 tiles, 256 threads, both global sides coalesced.

__global__ __launch_bounds__(256) void cvt_transpose_kernel(const float* __restrict__ in,
                                                            __bf16* __restrict__ out) {
  __shared__ __bf16 Ls[64][72];          // padded
  const int tid = threadIdx.x;
  const int k0 = blockIdx.x * 64, n0 = blockIdx.y * 64;
#pragma unroll
  for (int i = 0; i < 16; ++i) {
    const int idx = tid * 16 + i;        // 0..4095
    const int r = idx >> 6, c = idx & 63;
    Ls[r][c] = (__bf16)in[(long)(k0 + r) * DD + n0 + c];
  }
  __syncthreads();
#pragma unroll
  for (int i = 0; i < 16; ++i) {
    const int idx = tid * 16 + i;
    const int r = idx >> 6, c = idx & 63;   // out row n0+r, col k0+c
    out[(long)(n0 + r) * DD + k0 + c] = Ls[c][r];
  }
}

// ---------------- Kernel 1: QKV projection + feature map -------------------
// C[16384,1024] = Xb @ W (+bias); q/k get elu(t)+1. q stored row-major,
// km & v stored TRANSPOSED as [n*H+h][d][S] for kernel 2's fragment loads.
// grid=(256,16,3), 128 thr.

__global__ __launch_bounds__(128) void qkv_proj_kernel(
    const __bf16* __restrict__ xb,
    const __bf16* __restrict__ wqT, const __bf16* __restrict__ wkT,
    const __bf16* __restrict__ wvT,
    const float* __restrict__ bq, const float* __restrict__ bk,
    const float* __restrict__ bv,
    __bf16* __restrict__ qm, __bf16* __restrict__ kmT, __bf16* __restrict__ vT) {
  __shared__ __bf16 Xt[64 * 32];     // [m][k]
  __shared__ __bf16 Wt[64 * 32];     // [n][k]  (column-major B)
  const int tid = threadIdx.x, lane = tid & 31, w = tid >> 5;
  const int m0 = blockIdx.x * 64, n0 = blockIdx.y * 64;
  const int which = blockIdx.z;
  const __bf16* W    = (which == 0) ? wqT : (which == 1) ? wkT : wvT;
  const float*  bias = (which == 0) ? bq  : (which == 1) ? bk  : bv;

  v8f acc[4];
#pragma unroll
  for (int t = 0; t < 4; ++t)
#pragma unroll
    for (int r = 0; r < 8; ++r) acc[t][r] = 0.f;

  for (int k0 = 0; k0 < DD; k0 += 32) {
#pragma unroll
    for (int j = 0; j < 2; ++j) {          // X tile 64x32: 256 uint4 chunks
      const int c = tid + 128 * j;
      const int idx = c * 8;
      const int r = idx >> 5, cc = idx & 31;
      *(uint4*)&Xt[idx] = *(const uint4*)&xb[(long)(m0 + r) * DD + k0 + cc];
    }
#pragma unroll
    for (int j = 0; j < 2; ++j) {          // W^T tile 64x32
      const int c = tid + 128 * j;
      const int idx = c * 8;
      const int r = idx >> 5, cc = idx & 31;
      *(uint4*)&Wt[idx] = *(const uint4*)&W[(long)(n0 + r) * DD + k0 + cc];
    }
    if (k0 + 32 < DD) {                    // prefetch next K tile
      __builtin_prefetch(&xb[(long)(m0 + (tid >> 1)) * DD + k0 + 32], 0, 1);
      __builtin_prefetch(&W[(long)(n0 + (tid >> 1)) * DD + k0 + 32], 0, 1);
    }
    __syncthreads();
    const v16bf a = frag_a_rm(Xt, 32, 16 * w, lane);
#pragma unroll
    for (int t = 0; t < 4; ++t) {
      const v16bf b = frag_b_cm(Wt, 32, 16 * t, lane);
      acc[t] = wmma_bf16(a, b, acc[t]);
    }
    __syncthreads();
  }

  const int g = (lane >> 4) & 1, nl = lane & 15;
  const int nseq = m0 >> 12;               // batch index (m0 multiple of 64)
  const int h = n0 >> 6;                   // head index  (HD == 64)
#pragma unroll
  for (int t = 0; t < 4; ++t)
#pragma unroll
    for (int r = 0; r < 8; ++r) {
      const int mrow = 16 * w + r + 8 * g;
      const int ncol = 16 * t + nl;        // == d within head
      float v = acc[t][r] + bias[n0 + ncol];
      if (which < 2) v = (v > 0.f) ? (v + 1.f) : __expf(v);  // elu(v)+1
      if (which == 0) {
        qm[(long)(m0 + mrow) * DD + n0 + ncol] = (__bf16)v;
      } else {
        const int l = (m0 + mrow) & (SS - 1);
        __bf16* dst = (which == 1) ? kmT : vT;
        dst[((long)(nseq * HH + h) * HD + ncol) * SS + l] = (__bf16)v;
      }
    }
}

// ---------------- Kernel 2: kv = V^T @ Km, ksum = sum_s km -----------------
// km/v given transposed: [nh][d][S]. grid = B*H blocks, 128 thr.
// kv[m,d] = sum_s v[s,m] * km[s,d]: A[m,s]=vT[m][s] (row-major),
// B[s,d]=km[s,d]=kmT[d][s] (column-major) -> both contiguous fragments.

__global__ __launch_bounds__(128) void kv_reduce_kernel(
    const __bf16* __restrict__ kmT, const __bf16* __restrict__ vT,
    float* __restrict__ kv, float* __restrict__ ksum) {
  __shared__ __bf16 Kt[64 * 32];     // [d][s]
  __shared__ __bf16 Vt[64 * 32];     // [m][s]
  __shared__ float  ks[64];
  const int tid = threadIdx.x, lane = tid & 31, w = tid >> 5;
  const int nh = blockIdx.x;
  const long base = (long)nh * HD * SS;

  if (tid < 64) ks[tid] = 0.f;
  v8f acc[4];
#pragma unroll
  for (int t = 0; t < 4; ++t)
#pragma unroll
    for (int r = 0; r < 8; ++r) acc[t][r] = 0.f;

  for (int s0 = 0; s0 < SS; s0 += 32) {
#pragma unroll
    for (int j = 0; j < 2; ++j) {          // 64x32 tiles, uint4 chunks
      const int c = tid + 128 * j;
      const int idx = c * 8;
      const int d = idx >> 5, cc = idx & 31;
      const long src = base + (long)d * SS + s0 + cc;
      *(uint4*)&Kt[idx] = *(const uint4*)&kmT[src];
      *(uint4*)&Vt[idx] = *(const uint4*)&vT[src];
    }
    if (s0 + 32 < SS) {
      __builtin_prefetch(&kmT[base + (long)(tid >> 1) * SS + s0 + 32], 0, 1);
      __builtin_prefetch(&vT[base + (long)(tid >> 1) * SS + s0 + 32], 0, 1);
    }
    __syncthreads();
    if (tid < 64) {                        // ksum over this s-tile (row of Kt)
      float s = 0.f;
#pragma unroll
      for (int kk = 0; kk < 32; ++kk) s += (float)Kt[tid * 32 + kk];
      ks[tid] += s;
    }
    const v16bf a = frag_a_rm(Vt, 32, 16 * w, lane);
#pragma unroll
    for (int t = 0; t < 4; ++t) {
      const v16bf b = frag_b_cm(Kt, 32, 16 * t, lane);
      acc[t] = wmma_bf16(a, b, acc[t]);
    }
    __syncthreads();
  }

  const int g = (lane >> 4) & 1, nl = lane & 15;
#pragma unroll
  for (int t = 0; t < 4; ++t)
#pragma unroll
    for (int r = 0; r < 8; ++r) {
      const int m = 16 * w + r + 8 * g, d = 16 * t + nl;
      kv[((long)nh * 64 + m) * 64 + d] = acc[t][r];
    }
  if (tid < 64) ksum[nh * 64 + tid] = ks[tid];
}

// ---------------- Kernel 3: y = (Qm @ kv^T) * z ----------------------------
// grid = (S/64, B*H); z = 1/(qm . ksum + eps) per row.
// y[l,m] = sum_d qm[l,d]*kv[m,d]: A[l,d]=Qt row-major, B[d,m]=kv[m,d] =>
// Bt[m][d] row-major acts as column-major B -> contiguous fragments.

__global__ __launch_bounds__(128) void attn_y_kernel(
    const __bf16* __restrict__ qm, const float* __restrict__ kv,
    const float* __restrict__ ksum, __bf16* __restrict__ yb) {
  __shared__ __bf16 Qt[64 * 64];     // [l][d]
  __shared__ __bf16 Bt[64 * 64];     // [m][d]
  __shared__ float  ks[64];
  __shared__ float  zr[64];
  const int tid = threadIdx.x, lane = tid & 31, w = tid >> 5;
  const int lt = blockIdx.x, nh = blockIdx.y;
  const int n = nh >> 4, h = nh & 15;
  const long qbase = ((long)n * SS + lt * 64) * DD + h * HD;

#pragma unroll
  for (int j = 0; j < 4; ++j) {            // Qm tile 64x64: 512 uint4 chunks
    const int c = tid + 128 * j;
    const int idx = c * 8;
    const int r = idx >> 6, cc = idx & 63;
    *(uint4*)&Qt[idx] = *(const uint4*)&qm[qbase + (long)r * DD + cc];
  }
#pragma unroll
  for (int j = 0; j < 8; ++j) {            // kv 64x64 f32 -> bf16, row-major
    const int c = tid + 128 * j;           // 1024 f4 chunks
    const int idx = c * 4;
    const f4 v = *(const f4*)&kv[(long)nh * 4096 + idx];
    v4bf o;
    o[0] = (__bf16)v[0]; o[1] = (__bf16)v[1];
    o[2] = (__bf16)v[2]; o[3] = (__bf16)v[3];
    *(v4bf*)&Bt[idx] = o;
  }
  if (tid < 64) ks[tid] = ksum[nh * 64 + tid];
  __syncthreads();
  if (tid < 64) {                          // normalizer per row
    float s = 0.f;
#pragma unroll
    for (int d = 0; d < 64; ++d) s += (float)Qt[tid * 64 + d] * ks[d];
    zr[tid] = 1.f / (s + EPSV);
  }
  __syncthreads();

  v8f acc[4];
#pragma unroll
  for (int t = 0; t < 4; ++t)
#pragma unroll
    for (int r = 0; r < 8; ++r) acc[t][r] = 0.f;

#pragma unroll
  for (int k0 = 0; k0 < 64; k0 += 32) {
    const v16bf a = frag_a_rm(Qt + k0, 64, 16 * w, lane);
#pragma unroll
    for (int t = 0; t < 4; ++t) {
      const v16bf b = frag_b_cm(Bt + k0, 64, 16 * t, lane);
      acc[t] = wmma_bf16(a, b, acc[t]);
    }
  }

  const int g = (lane >> 4) & 1, nl = lane & 15;
#pragma unroll
  for (int t = 0; t < 4; ++t)
#pragma unroll
    for (int r = 0; r < 8; ++r) {
      const int row = 16 * w + r + 8 * g, col = 16 * t + nl;
      yb[qbase + (long)row * DD + col] = (__bf16)(acc[t][r] * zr[row]);
    }
}

// ---------------- Kernel 4: output projection ------------------------------
// out[16384,1024] = Y @ Wo + bo (fp32 out); Wo pre-transposed. grid=(256,16).

__global__ __launch_bounds__(128) void out_proj_kernel(
    const __bf16* __restrict__ yb, const __bf16* __restrict__ woT,
    const float* __restrict__ bo, float* __restrict__ out) {
  __shared__ __bf16 Yt[64 * 32];     // [m][k]
  __shared__ __bf16 Wt[64 * 32];     // [n][k]
  const int tid = threadIdx.x, lane = tid & 31, w = tid >> 5;
  const int m0 = blockIdx.x * 64, n0 = blockIdx.y * 64;

  v8f acc[4];
#pragma unroll
  for (int t = 0; t < 4; ++t)
#pragma unroll
    for (int r = 0; r < 8; ++r) acc[t][r] = 0.f;

  for (int k0 = 0; k0 < DD; k0 += 32) {
#pragma unroll
    for (int j = 0; j < 2; ++j) {
      const int c = tid + 128 * j;
      const int idx = c * 8;
      const int r = idx >> 5, cc = idx & 31;
      *(uint4*)&Yt[idx] = *(const uint4*)&yb[(long)(m0 + r) * DD + k0 + cc];
      *(uint4*)&Wt[idx] = *(const uint4*)&woT[(long)(n0 + r) * DD + k0 + cc];
    }
    if (k0 + 32 < DD) {
      __builtin_prefetch(&yb[(long)(m0 + (tid >> 1)) * DD + k0 + 32], 0, 1);
      __builtin_prefetch(&woT[(long)(n0 + (tid >> 1)) * DD + k0 + 32], 0, 1);
    }
    __syncthreads();
    const v16bf a = frag_a_rm(Yt, 32, 16 * w, lane);
#pragma unroll
    for (int t = 0; t < 4; ++t) {
      const v16bf b = frag_b_cm(Wt, 32, 16 * t, lane);
      acc[t] = wmma_bf16(a, b, acc[t]);
    }
    __syncthreads();
  }

  const int g = (lane >> 4) & 1, nl = lane & 15;
#pragma unroll
  for (int t = 0; t < 4; ++t)
#pragma unroll
    for (int r = 0; r < 8; ++r) {
      const int mrow = 16 * w + r + 8 * g;
      const int ncol = 16 * t + nl;
      out[(long)(m0 + mrow) * DD + n0 + ncol] = acc[t][r] + bo[n0 + ncol];
    }
}

// ---------------- Launch ---------------------------------------------------

extern "C" void kernel_launch(void* const* d_in, const int* in_sizes, int n_in,
                              void* d_out, int out_size, void* d_ws, size_t ws_size,
                              hipStream_t stream) {
  const float* x  = (const float*)d_in[0];
  const float* wq = (const float*)d_in[1];
  const float* bq = (const float*)d_in[2];
  const float* wk = (const float*)d_in[3];
  const float* bk = (const float*)d_in[4];
  const float* wv = (const float*)d_in[5];
  const float* bv = (const float*)d_in[6];
  const float* wo = (const float*)d_in[7];
  const float* bo = (const float*)d_in[8];
  float* out = (float*)d_out;

  const size_t NX = (size_t)NTOK * DD;   // 16.77M elems
  const size_t NW = (size_t)DD * DD;     // 1.05M elems

  // Workspace layout (256B aligned). yb aliases xb (xb dead after kernel 1).
  char* p = (char*)d_ws;
  size_t off = 0;
  auto take = [&](size_t bytes) {
    char* q = p + off;
    off = (off + bytes + 255) & ~(size_t)255;
    return q;
  };
  __bf16* xb  = (__bf16*)take(NX * 2);
  __bf16* wqT = (__bf16*)take(NW * 2);
  __bf16* wkT = (__bf16*)take(NW * 2);
  __bf16* wvT = (__bf16*)take(NW * 2);
  __bf16* woT = (__bf16*)take(NW * 2);
  __bf16* qmb = (__bf16*)take(NX * 2);
  __bf16* kmT = (__bf16*)take(NX * 2);
  __bf16* vT  = (__bf16*)take(NX * 2);
  float*  kvb = (float*)take((size_t)BB * HH * 64 * 64 * 4);
  float*  ksb = (float*)take((size_t)BB * HH * 64 * 4);
  __bf16* ybb = xb;                      // alias

  // 0) conversions: x plain; weights converted AND transposed to [N][K]
  cvt_bf16_kernel<<<1024, 256, 0, stream>>>(x, xb, (int)(NX / 4));
  cvt_transpose_kernel<<<dim3(16, 16), 256, 0, stream>>>(wq, wqT);
  cvt_transpose_kernel<<<dim3(16, 16), 256, 0, stream>>>(wk, wkT);
  cvt_transpose_kernel<<<dim3(16, 16), 256, 0, stream>>>(wv, wvT);
  cvt_transpose_kernel<<<dim3(16, 16), 256, 0, stream>>>(wo, woT);

  // 1) QKV projections (+ feature map on q,k); km/v written [nh][d][S]
  qkv_proj_kernel<<<dim3(NTOK / 64, DD / 64, 3), 128, 0, stream>>>(
      xb, wqT, wkT, wvT, bq, bk, bv, qmb, kmT, vT);

  // 2) kv summary + ksum per (n,h)
  kv_reduce_kernel<<<BB * HH, 128, 0, stream>>>(kmT, vT, kvb, ksb);

  // 3) y = (Qm @ kv^T) * z
  attn_y_kernel<<<dim3(SS / 64, BB * HH), 128, 0, stream>>>(qmb, kvb, ksb, ybb);

  // 4) output projection
  out_proj_kernel<<<dim3(NTOK / 64, DD / 64), 128, 0, stream>>>(ybb, woT, bo, out);
}
